// BandedHmmLm_30459908063247
// MI455X (gfx1250) — matrix operand order repaired and checked by reading, not compiled, not run on hardware
//
#include <hip/hip_runtime.h>

#define C_ 1024
#define H_ 256
#define V_ 10000
#define KBAND 32
#define K1K 65
#define B_ 8
#define T_ 256

typedef __attribute__((ext_vector_type(16))) __bf16 v16bf;
typedef __attribute__((ext_vector_type(8)))  float  v8f;

__device__ inline __bf16 f2bf(float f) {
  unsigned u = __builtin_bit_cast(unsigned, f);
  unsigned r = (u + 0x7fffu + ((u >> 16) & 1u)) >> 16;
  unsigned short s = (unsigned short)r;
  return __builtin_bit_cast(__bf16, s);
}

// A-matrix 16x32 bf16 tile loader (M x K), row-major source, ld in elements.
// ISA layout: lanes 0-15 (row m=lane): K 0-7 in comps 0-7, K 16-23 in comps 8-15;
//             lanes 16-31 (row m=lane-16): K 8-15 / K 24-31.
__device__ inline v16bf load_a16(const __bf16* base, int ld, int row0, int k0) {
  int lane = threadIdx.x & 31, half = lane >> 4, m = lane & 15;
  const __bf16* p = base + (size_t)(row0 + m) * ld + k0 + half * 8;
  v16bf v;
#pragma unroll
  for (int i = 0; i < 8; ++i) { v[i] = p[i]; v[i + 8] = p[i + 16]; }
  return v;
}

// B-matrix 32x16 bf16 tile loader (K x N). Source holds B^T row-major:
// element (k, n) at base[(col0+n)*ld + k]. Lanes 0-15: K 0-7 / K 16-23; lanes 16-31: K 8-15 / K 24-31.
__device__ inline v16bf load_b16(const __bf16* base, int ld, int col0, int k0) {
  int lane = threadIdx.x & 31, half = lane >> 4, n = lane & 15;
  const __bf16* p = base + (size_t)(col0 + n) * ld + k0 + half * 8;
  v16bf v;
#pragma unroll
  for (int i = 0; i < 8; ++i) { v[i] = p[i]; v[i + 8] = p[i + 16]; }
  return v;
}

__device__ inline v8f wmma_bf16(v16bf a, v16bf b, v8f c) {
  return __builtin_amdgcn_wmma_f32_16x16x32_bf16(false, a, false, b, (short)0, c,
                                                 false, false);
}

// Preload all A fragments for a K-chain into a register array (explicit residency:
// KTOT/32 tiles = KTOT/4 VGPRs; 64 VGPRs for K=256).
template <int KTOT>
__device__ inline void load_a_tiles(v16bf* at, const __bf16* __restrict__ Abase, int Ald,
                                    int row0, int koff = 0) {
#pragma unroll
  for (int k = 0; k < KTOT / 32; ++k) at[k] = load_a16(Abase, Ald, row0, koff + 32 * k);
}

// K-accumulating 16x16 tile GEMM: A from registers, B loads single-tile look-ahead
// so the next fetch is in flight while the current WMMA executes.
template <int KTOT>
__device__ inline v8f gemm_tile_a(const v16bf* at, const __bf16* __restrict__ Bbase, int Bld,
                                  int col0, v8f acc, int koff = 0) {
  v16bf b_cur = load_b16(Bbase, Bld, col0, koff);
#pragma unroll
  for (int k0 = 0; k0 < KTOT; k0 += 32) {
    v16bf b_nxt = b_cur;
    if (k0 + 32 < KTOT)  // compile-time under full unroll
      b_nxt = load_b16(Bbase, Bld, col0, koff + k0 + 32);
    acc = wmma_bf16(at[k0 / 32], b_cur, acc);
    b_cur = b_nxt;
  }
  return acc;
}

// Reductions across one 16-lane half (masks <16 stay within the half).
__device__ inline float hmax16(float v) {
#pragma unroll
  for (int m = 8; m; m >>= 1) v = fmaxf(v, __shfl_xor(v, m, 32));
  return v;
}
__device__ inline float hsum16(float v) {
#pragma unroll
  for (int m = 8; m; m >>= 1) v += __shfl_xor(v, m, 32);
  return v;
}
__device__ inline float wmax32(float v) {
#pragma unroll
  for (int m = 16; m; m >>= 1) v = fmaxf(v, __shfl_xor(v, m, 32));
  return v;
}
__device__ inline float wsum32(float v) {
#pragma unroll
  for (int m = 16; m; m >>= 1) v += __shfl_xor(v, m, 32);
  return v;
}

// Branchless banded-logit addition: clamp diagonal offset into the valid band so
// the load is always in-bounds, then select with cndmask (no exec-mask branch).
__device__ inline float band_add(const float* __restrict__ band, int i, int j, float x) {
  int d = j - i;
  int dc = d < -KBAND ? -KBAND : (d > KBAND ? KBAND : d);
  float bv = band[i * K1K + dc + KBAND];
  return x + ((d == dc) ? bv : 0.f);
}

// ---------------------------------------------------------------- converts
__global__ void k_f32_to_bf16(const float* __restrict__ in, __bf16* __restrict__ out, int n) {
  int i = blockIdx.x * blockDim.x + threadIdx.x;
  if (i < n) out[i] = f2bf(in[i]);
}

// ---------------------------------------------------------------- start head (tiny, VALU)
__global__ void k_start(const float* __restrict__ start_emb, const float* __restrict__ linW,
                        const float* __restrict__ linb, const float* __restrict__ W1,
                        const float* __restrict__ b1, const float* __restrict__ W2,
                        const float* __restrict__ b2, const float* __restrict__ next_emb,
                        float* __restrict__ startv) {
  __shared__ float x[H_], tmp[H_], tmp2[H_], logits[C_], red[256];
  int t = threadIdx.x;
  x[t] = start_emb[t];
  __syncthreads();
  float a0 = linb[t];
  for (int k = 0; k < H_; ++k) a0 += x[k] * linW[t * H_ + k];
  tmp[t] = a0;
  __syncthreads();
  x[t] = tmp[t];
  __syncthreads();
  for (int l = 0; l < 2; ++l) {
    const float* w1 = W1 + l * H_ * H_;
    const float* w2 = W2 + l * H_ * H_;
    float a = b1[l * H_ + t];
    for (int k = 0; k < H_; ++k) a += x[k] * w1[t * H_ + k];
    tmp[t] = fmaxf(a, 0.f);
    __syncthreads();
    float c = b2[l * H_ + t];
    for (int k = 0; k < H_; ++k) c += tmp[k] * w2[t * H_ + k];
    tmp2[t] = fmaxf(c, 0.f);
    __syncthreads();
    x[t] += tmp2[t];
    __syncthreads();
  }
  for (int q = 0; q < 4; ++q) {
    int c = t + 256 * q;
    const float* e = next_emb + (size_t)c * H_;
    float a = 0.f;
    for (int k = 0; k < H_; ++k) a += x[k] * e[k];
    logits[c] = a;
  }
  __syncthreads();
  float m = -1e30f;
  for (int q = 0; q < 4; ++q) m = fmaxf(m, logits[t + 256 * q]);
  red[t] = m;
  __syncthreads();
  for (int s = 128; s; s >>= 1) {
    if (t < s) red[t] = fmaxf(red[t], red[t + s]);
    __syncthreads();
  }
  m = red[0];
  __syncthreads();
  float su = 0.f;
  for (int q = 0; q < 4; ++q) su += expf(logits[t + 256 * q] - m);
  red[t] = su;
  __syncthreads();
  for (int s = 128; s; s >>= 1) {
    if (t < s) red[t] += red[t + s];
    __syncthreads();
  }
  float lse = m + logf(red[0]);
  for (int q = 0; q < 4; ++q) {
    int c = t + 256 * q;
    startv[c] = logits[c] - lse;
  }
}

// ---------------------------------------------------------------- terminal ResLayer MLP (VALU)
__global__ void k_term_mlp(const float* __restrict__ pre, const float* __restrict__ W1,
                           const float* __restrict__ b1, const float* __restrict__ W2,
                           const float* __restrict__ b2, __bf16* __restrict__ ft_bf) {
  __shared__ float x[H_], h[H_], h2[H_];
  int t = threadIdx.x;
  int row = blockIdx.x;
  x[t] = pre[(size_t)row * H_ + t];
  __syncthreads();
  for (int l = 0; l < 2; ++l) {
    const float* w1 = W1 + l * H_ * H_;
    const float* w2 = W2 + l * H_ * H_;
    float a = b1[l * H_ + t];
    for (int k = 0; k < H_; ++k) a += x[k] * w1[t * H_ + k];
    h[t] = fmaxf(a, 0.f);
    __syncthreads();
    float c = b2[l * H_ + t];
    for (int k = 0; k < H_; ++k) c += h[k] * w2[t * H_ + k];
    h2[t] = fmaxf(c, 0.f);
    __syncthreads();
    x[t] += h2[t];
    __syncthreads();
  }
  ft_bf[(size_t)row * H_ + t] = f2bf(x[t]);
}

// ---------------------------------------------------------------- transition: WMMA + band +
// two-pass online row-softmax; writes P^T = exp(log_softmax(logits+band)) transposed, bf16.
__global__ void k_transition(const __bf16* __restrict__ sbf, const __bf16* __restrict__ nbf,
                             const float* __restrict__ band, __bf16* __restrict__ Pt) {
  __shared__ float pm[8 * 16], ps[8 * 16], rowL[16];
  int rb = blockIdx.x * 16;
  int wave = threadIdx.x >> 5, lane = threadIdx.x & 31, half = lane >> 4, n = lane & 15;
  v16bf at[H_ / 32];
  load_a_tiles<H_>(at, sbf, H_, rb);  // 16 rows of state_emb, resident in 64 VGPRs
  float M[8], S[8];
#pragma unroll
  for (int r = 0; r < 8; ++r) { M[r] = -1e30f; S[r] = 0.f; }
  for (int q = 0; q < 8; ++q) {
    int cb = (wave + 8 * q) * 16;
    v8f acc = {0.f, 0.f, 0.f, 0.f, 0.f, 0.f, 0.f, 0.f};
    acc = gemm_tile_a<H_>(at, nbf, H_, cb, acc);
#pragma unroll
    for (int r = 0; r < 8; ++r) {
      int i = rb + r + 8 * half, j = cb + n;
      float x = band_add(band, i, j, acc[r]);
      float tm = hmax16(x);
      float ts = hsum16(expf(x - tm));
      float nm = fmaxf(M[r], tm);
      S[r] = S[r] * expf(M[r] - nm) + ts * expf(tm - nm);
      M[r] = nm;
    }
  }
  if (n == 0) {
#pragma unroll
    for (int r = 0; r < 8; ++r) {
      pm[wave * 16 + 8 * half + r] = M[r];
      ps[wave * 16 + 8 * half + r] = S[r];
    }
  }
  __syncthreads();
  if (threadIdx.x < 16) {
    int row = threadIdx.x;
    float mm = -1e30f;
    for (int w = 0; w < 8; ++w) mm = fmaxf(mm, pm[w * 16 + row]);
    float ss = 0.f;
    for (int w = 0; w < 8; ++w) ss += ps[w * 16 + row] * expf(pm[w * 16 + row] - mm);
    rowL[row] = mm + logf(ss);
  }
  __syncthreads();
  // pass 2: recompute tiles (A still in registers), emit transposed probabilities
  for (int q = 0; q < 8; ++q) {
    int cb = (wave + 8 * q) * 16;
    v8f acc = {0.f, 0.f, 0.f, 0.f, 0.f, 0.f, 0.f, 0.f};
    acc = gemm_tile_a<H_>(at, nbf, H_, cb, acc);
#pragma unroll
    for (int r = 0; r < 8; ++r) {
      int m = r + 8 * half;
      int i = rb + m, j = cb + n;
      float x = band_add(band, i, j, acc[r]);
      Pt[(size_t)j * C_ + i] = f2bf(expf(x - rowL[m]));
    }
  }
}

// ---------------------------------------------------------------- emission normalizers:
// lse[c] = logsumexp_v ( ft[c] . term[v] ), online over 625 WMMA tiles.
__global__ void k_emit_lse(const __bf16* __restrict__ ft_bf, const __bf16* __restrict__ term_bf,
                           float* __restrict__ lse) {
  __shared__ float pm[8 * 16], ps[8 * 16];
  int rb = blockIdx.x * 16;
  int wave = threadIdx.x >> 5, lane = threadIdx.x & 31, half = lane >> 4, n = lane & 15;
  v16bf at[H_ / 32];
  load_a_tiles<H_>(at, ft_bf, H_, rb);  // 16 rows of ft, resident in 64 VGPRs
  float M[8], S[8];
#pragma unroll
  for (int r = 0; r < 8; ++r) { M[r] = -1e30f; S[r] = 0.f; }
  for (int vt = wave; vt < V_ / 16; vt += 8) {
    int vn = vt + 8;  // prefetch next tile of terminal embeddings for this lane
    if (vn < V_ / 16)
      __builtin_prefetch(term_bf + (size_t)(vn * 16 + n) * H_ + half * 16, 0, 0);
    int vb = vt * 16;
    v8f acc = {0.f, 0.f, 0.f, 0.f, 0.f, 0.f, 0.f, 0.f};
    acc = gemm_tile_a<H_>(at, term_bf, H_, vb, acc);
#pragma unroll
    for (int r = 0; r < 8; ++r) {
      float x = acc[r];
      float tm = hmax16(x);
      float ts = hsum16(expf(x - tm));
      float nm = fmaxf(M[r], tm);
      S[r] = S[r] * expf(M[r] - nm) + ts * expf(tm - nm);
      M[r] = nm;
    }
  }
  if (n == 0) {
#pragma unroll
    for (int r = 0; r < 8; ++r) {
      pm[wave * 16 + 8 * half + r] = M[r];
      ps[wave * 16 + 8 * half + r] = S[r];
    }
  }
  __syncthreads();
  if (threadIdx.x < 16) {
    int row = threadIdx.x;
    float mm = -1e30f;
    for (int w = 0; w < 8; ++w) mm = fmaxf(mm, pm[w * 16 + row]);
    float ss = 0.f;
    for (int w = 0; w < 8; ++w) ss += ps[w * 16 + row] * expf(pm[w * 16 + row] - mm);
    lse[rb + row] = mm + logf(ss);
  }
}

// ---------------------------------------------------------------- gather used-token embeddings
__global__ void k_gather_tok(const int* __restrict__ text, const float* __restrict__ term_emb,
                             __bf16* __restrict__ tokA) {
  int p = blockIdx.x;
  int t = threadIdx.x;
  int tok = text[p];
  tokA[(size_t)p * H_ + t] = f2bf(term_emb[(size_t)tok * H_ + t]);
}

// emit[p, c] = ft[c] . term[text[p]] - lse[c]   (log emission prob of used tokens)
__global__ void k_emit_used(const __bf16* __restrict__ tokA, const __bf16* __restrict__ ft_bf,
                            const float* __restrict__ lse, float* __restrict__ emit) {
  int cb = blockIdx.x * 16;
  int pb = blockIdx.y * 16;
  int lane = threadIdx.x & 31, half = lane >> 4, n = lane & 15;
  v16bf at[H_ / 32];
  load_a_tiles<H_>(at, tokA, H_, pb);
  v8f acc = {0.f, 0.f, 0.f, 0.f, 0.f, 0.f, 0.f, 0.f};
  acc = gemm_tile_a<H_>(at, ft_bf, H_, cb, acc);
  float l = lse[cb + n];
#pragma unroll
  for (int r = 0; r < 8; ++r) {
    int p = pb + r + 8 * half;
    emit[(size_t)p * C_ + cb + n] = acc[r] - l;
  }
}

// ---------------------------------------------------------------- forward scan
__global__ void k_alpha_init(const float* __restrict__ startv, const float* __restrict__ emit,
                             float* __restrict__ alpha, __bf16* __restrict__ a_bf) {
  int g = blockIdx.x * blockDim.x + threadIdx.x;  // 0..16383
  int b = g >> 10, c = g & (C_ - 1);
  float v;
  if (b < B_) {
    v = startv[c] + emit[((size_t)b * T_ + 0) * C_ + c];
  } else {
    v = -1e30f;
    a_bf[g] = f2bf(0.f);  // pad rows stay zero in linear space forever
  }
  alpha[g] = v;
}

// per-row max-normalize, emit exp(alpha - M) in bf16
__global__ void k_alpha_norm(const float* __restrict__ alpha, __bf16* __restrict__ a_bf,
                             float* __restrict__ Mrow) {
  int b = threadIdx.x >> 5, lane = threadIdx.x & 31;
  float vals[32];
  float m = -1e30f;
#pragma unroll
  for (int k = 0; k < 32; ++k) {
    vals[k] = alpha[(size_t)b * C_ + lane + 32 * k];
    m = fmaxf(m, vals[k]);
  }
  m = wmax32(m);
#pragma unroll
  for (int k = 0; k < 32; ++k)
    a_bf[(size_t)b * C_ + lane + 32 * k] = f2bf(expf(vals[k] - m));
  if (lane == 0) Mrow[b] = m;
}

// alpha'[b, j] = M_b + log( sum_i exp(alpha-M)[b,i] * P[i,j] ) + emit[b*T+t, j]
// K=1024 split across 4 waves (256 each, 8 WMMAs) -> 4x shorter dependent WMMA
// chain per sequential scan step; partial tiles summed through LDS.
__global__ void k_alpha_step(const __bf16* __restrict__ a_bf, const __bf16* __restrict__ Pt,
                             const float* __restrict__ Mrow, const float* __restrict__ emit,
                             int tstep, float* __restrict__ alpha_out) {
  __shared__ float pa[4 * 256];
  int cb = blockIdx.x * 16;
  int wave = threadIdx.x >> 5, lane = threadIdx.x & 31, half = lane >> 4, n = lane & 15;
  v16bf at[H_ / 32];
  load_a_tiles<H_>(at, a_bf, C_, 0, wave * (C_ / 4));  // this wave's K-slice of alpha
  v8f acc = {0.f, 0.f, 0.f, 0.f, 0.f, 0.f, 0.f, 0.f};
  acc = gemm_tile_a<C_ / 4>(at, Pt, C_, cb, acc, wave * (C_ / 4));
#pragma unroll
  for (int r = 0; r < 8; ++r)
    pa[wave * 256 + (r + 8 * half) * 16 + n] = acc[r];
  __syncthreads();
  int t = threadIdx.x;  // 0..127
#pragma unroll
  for (int e = t; e < 256; e += 128) {
    int m = e >> 4, n2 = e & 15;
    float s = pa[e] + pa[256 + e] + pa[512 + e] + pa[768 + e];
    int j = cb + n2;
    float o;
    if (m < B_)
      o = Mrow[m] + logf(fmaxf(s, 1e-37f)) + emit[((size_t)m * T_ + tstep) * C_ + j];
    else
      o = -1e30f;
    alpha_out[(size_t)m * C_ + j] = o;
  }
}

__global__ void k_final(const float* __restrict__ alpha, float* __restrict__ out) {
  int b = threadIdx.x >> 5, lane = threadIdx.x & 31;
  float vals[32];
  float m = -1e30f;
#pragma unroll
  for (int k = 0; k < 32; ++k) {
    vals[k] = alpha[(size_t)b * C_ + lane + 32 * k];
    m = fmaxf(m, vals[k]);
  }
  m = wmax32(m);
  float s = 0.f;
#pragma unroll
  for (int k = 0; k < 32; ++k) s += expf(vals[k] - m);
  s = wsum32(s);
  if (lane == 0) out[b] = m + logf(s);
}

// ================================================================ host
extern "C" void kernel_launch(void* const* d_in, const int* in_sizes, int n_in,
                              void* d_out, int out_size, void* d_ws, size_t ws_size,
                              hipStream_t stream) {
  (void)in_sizes; (void)n_in; (void)out_size; (void)ws_size;
  const int*   text      = (const int*)d_in[0];
  const float* start_emb = (const float*)d_in[1];
  const float* slinW     = (const float*)d_in[2];
  const float* slinb     = (const float*)d_in[3];
  const float* sW1       = (const float*)d_in[4];
  const float* sb1       = (const float*)d_in[5];
  const float* sW2       = (const float*)d_in[6];
  const float* sb2       = (const float*)d_in[7];
  const float* state_emb = (const float*)d_in[8];
  const float* next_emb  = (const float*)d_in[9];
  const float* pre_emb   = (const float*)d_in[10];
  const float* tW1       = (const float*)d_in[11];
  const float* tb1       = (const float*)d_in[12];
  const float* tW2       = (const float*)d_in[13];
  const float* tb2       = (const float*)d_in[14];
  const float* term_emb  = (const float*)d_in[15];
  const float* band      = (const float*)d_in[16];

  char* ws = (char*)d_ws;
  size_t off = 0;
  auto alloc = [&](size_t bytes) -> void* {
    void* p = ws + off;
    off += (bytes + 255) & ~(size_t)255;
    return p;
  };
  __bf16* sbf    = (__bf16*)alloc((size_t)C_ * H_ * 2);
  __bf16* nbf    = (__bf16*)alloc((size_t)C_ * H_ * 2);
  __bf16* tbf    = (__bf16*)alloc((size_t)V_ * H_ * 2);
  __bf16* ftbf   = (__bf16*)alloc((size_t)C_ * H_ * 2);
  __bf16* Pt     = (__bf16*)alloc((size_t)C_ * C_ * 2);
  float*  lse    = (float*)alloc((size_t)C_ * 4);
  float*  startv = (float*)alloc((size_t)C_ * 4);
  __bf16* tokA   = (__bf16*)alloc((size_t)B_ * T_ * H_ * 2);
  float*  emit   = (float*)alloc((size_t)B_ * T_ * C_ * 4);
  float*  alphaA = (float*)alloc((size_t)16 * C_ * 4);
  float*  alphaB = (float*)alloc((size_t)16 * C_ * 4);
  __bf16* a_bf   = (__bf16*)alloc((size_t)16 * C_ * 2);
  float*  Mrow   = (float*)alloc(16 * 4);

  int n1 = C_ * H_;
  k_f32_to_bf16<<<(n1 + 255) / 256, 256, 0, stream>>>(state_emb, sbf, n1);
  k_f32_to_bf16<<<(n1 + 255) / 256, 256, 0, stream>>>(next_emb, nbf, n1);
  int n2 = V_ * H_;
  k_f32_to_bf16<<<(n2 + 255) / 256, 256, 0, stream>>>(term_emb, tbf, n2);

  k_start<<<1, 256, 0, stream>>>(start_emb, slinW, slinb, sW1, sb1, sW2, sb2, next_emb, startv);
  k_term_mlp<<<C_, 256, 0, stream>>>(pre_emb, tW1, tb1, tW2, tb2, ftbf);
  k_transition<<<C_ / 16, 256, 0, stream>>>(sbf, nbf, band, Pt);
  k_emit_lse<<<C_ / 16, 256, 0, stream>>>(ftbf, tbf, lse);
  k_gather_tok<<<B_ * T_, 256, 0, stream>>>(text, term_emb, tokA);
  k_emit_used<<<dim3(C_ / 16, (B_ * T_) / 16), 32, 0, stream>>>(tokA, ftbf, lse, emit);

  k_alpha_init<<<64, 256, 0, stream>>>(startv, emit, alphaA, a_bf);
  float* cur = alphaA;
  float* nxt = alphaB;
  for (int t = 1; t < T_; ++t) {
    k_alpha_norm<<<1, 256, 0, stream>>>(cur, a_bf, Mrow);
    k_alpha_step<<<C_ / 16, 128, 0, stream>>>(a_bf, Pt, Mrow, emit, t, nxt);
    float* tmp = cur; cur = nxt; nxt = tmp;
  }
  k_final<<<1, 256, 0, stream>>>(cur, (float*)d_out);
}
